// GDNNaylisAttention_35545149342130
// MI455X (gfx1250) — compile-verified, matrix operand order checked by reading, not run on hardware
//
#include <hip/hip_runtime.h>

// ---------------------------------------------------------------- types
typedef __bf16 bf16;
typedef bf16  v16bf __attribute__((ext_vector_type(16)));
typedef bf16  v8bf  __attribute__((ext_vector_type(8)));
typedef float v8f   __attribute__((ext_vector_type(8)));

#define Bc 2
#define Sc 4096
#define Dc 1024
#define Hc 16
#define dc 64
#define Rc 48
#define HRc (Hc * Rc)   // 768
#define CHK 32          // scan chunk staged in LDS (double-buffered)

// ---------------------------------------------------------------- async copy (gfx1250)
// global_load_async_to_lds_b128: VDST = VGPR with LDS byte offset, VADDR = 64-bit
// global address. Tracked by ASYNCcnt; in-order completion among async loads.
static __device__ inline void async_copy_b128(const void* gptr, void* lptr) {
    unsigned int lds_off = (unsigned int)(unsigned long long)(uintptr_t)lptr;
    unsigned long long ga = (unsigned long long)(uintptr_t)gptr;
    asm volatile("global_load_async_to_lds_b128 %0, %1, off"
                 :: "v"(lds_off), "v"(ga) : "memory");
}
#define WAIT_ASYNCCNT(n) asm volatile("s_wait_asynccnt " #n ::: "memory")

// ---------------------------------------------------------------- helpers
static __device__ inline v16bf load_frag(const bf16* __restrict__ row, int klo) {
    union { v16bf v; v8bf h[2]; } u;
    u.h[0] = *reinterpret_cast<const v8bf*>(row + klo);
    u.h[1] = *reinterpret_cast<const v8bf*>(row + klo + 16);
    return u.v;
}

// ---------------------------------------------------------------- cast / transpose prep
__global__ __launch_bounds__(256) void cast_f32_bf16(const float* __restrict__ in,
                                                     bf16* __restrict__ out, int n) {
    int i = blockIdx.x * 256 + threadIdx.x;
    if (i < n) out[i] = (bf16)in[i];
}

// W: [K,N] row-major fp32  ->  WT: [N,K] row-major bf16
__global__ __launch_bounds__(256) void transpose_cast(const float* __restrict__ W,
                                                      bf16* __restrict__ WT,
                                                      int K, int N) {
    int n = blockIdx.x;
    for (int k = threadIdx.x; k < K; k += 256)
        WT[(size_t)n * K + k] = (bf16)W[(size_t)k * N + n];
}

// ---------------------------------------------------------------- WMMA GEMM
// C[M,N] = A[M,K] * B[K,N];  A row-major bf16, B given pre-transposed BT[N,K] bf16.
// Block: 256 threads = 8 waves. Wave computes a 32(M) x 64(N) patch with
// 2x4 v_wmma_f32_16x16x32_bf16 accumulators. Block tile: 256(M) x 64(N).
__global__ __launch_bounds__(256)
void wmma_gemm(const bf16* __restrict__ A, const bf16* __restrict__ BT,
               float* __restrict__ C, int M, int N, int Kd) {
    const int wave = threadIdx.x >> 5;
    const int lane = threadIdx.x & 31;
    const int m0 = blockIdx.y * 256 + wave * 32;
    const int n0 = blockIdx.x * 64;
    const int lrow = lane & 15;
    const int kofs = (lane >> 4) << 3;   // 0 for lanes 0-15, 8 for lanes 16-31

    v8f acc[2][4] = {};

    const bf16* a0p = A + (size_t)(m0 + lrow) * Kd;
    const bf16* a1p = A + (size_t)(m0 + 16 + lrow) * Kd;
    const bf16* bp[4];
#pragma unroll
    for (int j = 0; j < 4; ++j) bp[j] = BT + (size_t)(n0 + j * 16 + lrow) * Kd;

    for (int kb = 0; kb < Kd; kb += 32) {
        const int klo = kb + kofs;
        __builtin_prefetch(a0p + kb + 128, 0, 3);   // global_prefetch_b8
        v16bf af0 = load_frag(a0p, klo);
        v16bf af1 = load_frag(a1p, klo);
#pragma unroll
        for (int j = 0; j < 4; ++j) {
            v16bf bf = load_frag(bp[j], klo);
            acc[0][j] = __builtin_amdgcn_wmma_f32_16x16x32_bf16(
                false, af0, false, bf, (short)0, acc[0][j], false, false);
            acc[1][j] = __builtin_amdgcn_wmma_f32_16x16x32_bf16(
                false, af1, false, bf, (short)0, acc[1][j], false, false);
        }
    }

    const int rofs = (lane >> 4) << 3;
#pragma unroll
    for (int i = 0; i < 2; ++i)
#pragma unroll
        for (int j = 0; j < 4; ++j)
#pragma unroll
            for (int r = 0; r < 8; ++r) {
                int row = m0 + i * 16 + r + rofs;
                int col = n0 + j * 16 + lrow;
                C[(size_t)row * N + col] = acc[i][j][r];
            }
}

// ---------------------------------------------------------------- fused conv + norms
// P: [B,S,D] fp32 (projection). One block = 64 threads = one (b,s,h) head.
// mode 0: q -> rmsnorm*qn_w + lam*Rq            -> out
// mode 1: k -> L2-normalize(rmsnorm*kn_w+lam*Rk)-> out
// mode 2: v -> conv only                        -> out
// out layout: [B*H, S, d]
__global__ __launch_bounds__(64)
void fuse_qkv(const float* __restrict__ P, const float* __restrict__ cw,
              const float* __restrict__ nw, const float* __restrict__ Rlr,
              const float* __restrict__ gs, float* __restrict__ out, int mode) {
    const int s = blockIdx.x, h = blockIdx.y, b = blockIdx.z;
    const int t = threadIdx.x;
    const int c = h * dc + t;

    float acc = 0.f;
#pragma unroll
    for (int j = 0; j < 4; ++j) {
        int sp = s + j - 3;
        if (sp >= 0) acc += cw[c * 4 + j] * P[((size_t)b * Sc + sp) * Dc + c];
    }

    __shared__ float red[64];
    float val;
    if (mode == 2) {
        val = acc;
    } else {
        red[t] = acc * acc; __syncthreads();
#pragma unroll
        for (int st = 32; st > 0; st >>= 1) { if (t < st) red[t] += red[t + st]; __syncthreads(); }
        float ms = red[0] * (1.f / dc);
        __syncthreads();
        float nrmv = acc * rsqrtf(ms + 1e-6f) * nw[t];
        float lam = fabsf(gs[h]);
        float rl  = (t < Rc) ? Rlr[((size_t)b * Sc + s) * HRc + h * Rc + t] : 0.f;
        float u = nrmv + lam * rl;
        if (mode == 0) {
            val = u;
        } else {
            red[t] = u * u; __syncthreads();
#pragma unroll
            for (int st = 32; st > 0; st >>= 1) { if (t < st) red[t] += red[t + st]; __syncthreads(); }
            float l2 = sqrtf(red[0]);
            val = u / fmaxf(l2, 1e-12f);
        }
    }
    out[(((size_t)b * Hc + h) * Sc + s) * dc + t] = val;
}

// ---------------------------------------------------------------- gates alpha/beta
// alpha,beta: [B,H,S]
__global__ __launch_bounds__(256)
void gates_kernel(const float* __restrict__ x, const float* __restrict__ Wf,
                  const float* __restrict__ bfv, const float* __restrict__ Wl,
                  const float* __restrict__ blv, float* __restrict__ alpha,
                  float* __restrict__ beta) {
    const int s = blockIdx.x, b = blockIdx.y;
    __shared__ float xs[Dc];
    const float* xr = x + ((size_t)b * Sc + s) * Dc;
    for (int i = threadIdx.x; i < Dc; i += 256) xs[i] = xr[i];
    __syncthreads();

    const int o = threadIdx.x >> 3;   // 0..31  (0..15 -> alpha, 16..31 -> beta)
    const int sub = threadIdx.x & 7;
    const int oc = o & 15;
    const float* W = (o < Hc) ? Wf : Wl;
    float acc = 0.f;
    for (int i = sub; i < Dc; i += 8) acc += xs[i] * W[(size_t)i * Hc + oc];
    acc += __shfl_xor(acc, 1, 8);
    acc += __shfl_xor(acc, 2, 8);
    acc += __shfl_xor(acc, 4, 8);
    if (sub == 0) {
        float bias = (o < Hc) ? bfv[oc] : blv[oc];
        float g = 1.f / (1.f + __expf(-(acc + bias)));
        float* dst = (o < Hc) ? alpha : beta;
        dst[((size_t)b * Hc + oc) * Sc + s] = g;
    }
}

// ---------------------------------------------------------------- gated delta-rule scan
// One block per (b,h). 512 threads: row = tid/8 (64 state rows), sub = tid%8
// owns 8 contiguous columns of mem in registers. Width-8 shuffle reductions.
// k/v/q chunks are staged into LDS with double-buffered async copies
// (global_load_async_to_lds_b128 + s_wait_asynccnt) so the next chunk's fetch
// overlaps the strictly sequential recurrence.
__global__ __launch_bounds__(512)
void scan_kernel(const float* __restrict__ Kt, const float* __restrict__ Vt,
                 const float* __restrict__ Qt, const float* __restrict__ alpha,
                 const float* __restrict__ beta, float* __restrict__ O) {
    const int bh = blockIdx.x;
    const int b = bh / Hc, h = bh % Hc;
    const int tid = threadIdx.x;
    const int row = tid >> 3;
    const int sub = tid & 7;
    const int NC = Sc / CHK;
    const int CE = CHK * dc;          // 2048 floats per array per chunk

    float m[8];
#pragma unroll
    for (int j = 0; j < 8; ++j) m[j] = 0.f;

    __shared__ float smem[2][3 * CHK * dc];   // [buf][ k | v | q ]
    __shared__ float sab[2][2][CHK];          // [buf][a,b][t]

    // issue one chunk's async loads: 3 x b128 per thread (k,v,q), gates via LDS store
    auto issue = [&](int buf, int ck) {
        size_t base = ((size_t)bh * Sc + ck * CHK) * dc;
        int idx = tid * 4;            // 512 threads * 4 floats = 2048
        async_copy_b128(Kt + base + idx, &smem[buf][0 * CE + idx]);
        async_copy_b128(Vt + base + idx, &smem[buf][1 * CE + idx]);
        async_copy_b128(Qt + base + idx, &smem[buf][2 * CE + idx]);
        if (tid < CHK) {
            sab[buf][0][tid] = alpha[(size_t)bh * Sc + ck * CHK + tid];
            sab[buf][1][tid] = beta[(size_t)bh * Sc + ck * CHK + tid];
        }
    };

    issue(0, 0);                      // pending ASYNCcnt = 3

    for (int ck = 0; ck < NC; ++ck) {
        const int buf = ck & 1;
        if (ck + 1 < NC) {
            issue(buf ^ 1, ck + 1);   // pending = 6; in-order completion
            WAIT_ASYNCCNT(3);         // current chunk resident, next in flight
        } else {
            WAIT_ASYNCCNT(0);
        }
        __syncthreads();              // cross-wave visibility of LDS data

        const float* sk = &smem[buf][0 * CE];
        const float* sv = &smem[buf][1 * CE];
        const float* sq = &smem[buf][2 * CE];
        const float* sA = &sab[buf][0][0];
        const float* sB = &sab[buf][1][0];

        for (int t = 0; t < CHK; ++t) {
            const float* kp = sk + t * dc + sub * 8;
            const float* qp = sq + t * dc + sub * 8;
            float kv[8];
#pragma unroll
            for (int j = 0; j < 8; ++j) kv[j] = kp[j];

            float r = 0.f;
#pragma unroll
            for (int j = 0; j < 8; ++j) r += m[j] * kv[j];
            r += __shfl_xor(r, 1, 8);
            r += __shfl_xor(r, 2, 8);
            r += __shfl_xor(r, 4, 8);

            float a = sA[t], bg = sB[t];
            float delta = sv[t * dc + row] - r;
            float coef = bg * delta;
            float o = 0.f;
#pragma unroll
            for (int j = 0; j < 8; ++j) {
                m[j] = a * m[j] + coef * kv[j];
                o += m[j] * qp[j];
            }
            o += __shfl_xor(o, 1, 8);
            o += __shfl_xor(o, 2, 8);
            o += __shfl_xor(o, 4, 8);
            if (sub == 0) {
                int sidx = ck * CHK + t;
                O[(((size_t)b * Sc + sidx) * Hc + h) * dc + row] = o;
            }
        }
        __syncthreads();              // all waves done reading buf before reuse
    }
}

// ---------------------------------------------------------------- final rmsnorm -> bf16
__global__ __launch_bounds__(256)
void final_norm(const float* __restrict__ O, const float* __restrict__ onw,
                bf16* __restrict__ Yn) {
    const int s = blockIdx.x, b = blockIdx.y;
    const size_t off = ((size_t)b * Sc + s) * Dc;
    float v[4], ss = 0.f;
#pragma unroll
    for (int i = 0; i < 4; ++i) {
        v[i] = O[off + threadIdx.x + 256 * i];
        ss += v[i] * v[i];
    }
    __shared__ float red[256];
    red[threadIdx.x] = ss; __syncthreads();
#pragma unroll
    for (int st = 128; st > 0; st >>= 1) {
        if (threadIdx.x < st) red[threadIdx.x] += red[threadIdx.x + st];
        __syncthreads();
    }
    float scl = rsqrtf(red[0] * (1.f / Dc) + 1e-6f);
#pragma unroll
    for (int i = 0; i < 4; ++i) {
        int c = threadIdx.x + 256 * i;
        Yn[off + c] = (bf16)(v[i] * scl * onw[c]);
    }
}

// ---------------------------------------------------------------- host
extern "C" void kernel_launch(void* const* d_in, const int* in_sizes, int n_in,
                              void* d_out, int out_size, void* d_ws, size_t ws_size,
                              hipStream_t stream) {
    const float* x   = (const float*)d_in[0];
    const float* Wq  = (const float*)d_in[1];
    const float* Wk  = (const float*)d_in[2];
    const float* Wv  = (const float*)d_in[3];
    const float* Wo  = (const float*)d_in[4];
    const float* cq  = (const float*)d_in[5];
    const float* ckw = (const float*)d_in[6];
    const float* cv  = (const float*)d_in[7];
    const float* Wf  = (const float*)d_in[8];
    const float* bfv = (const float*)d_in[9];
    const float* Wl  = (const float*)d_in[10];
    const float* blv = (const float*)d_in[11];
    const float* Wrq = (const float*)d_in[12];
    const float* Wrk = (const float*)d_in[13];
    const float* gs  = (const float*)d_in[14];
    const float* qnw = (const float*)d_in[15];
    const float* knw = (const float*)d_in[16];
    const float* onw = (const float*)d_in[17];
    float* out = (float*)d_out;

    size_t off = 0;
    auto alloc = [&](size_t bytes) -> void* {
        void* p = (char*)d_ws + off;
        off = (off + bytes + 255) & ~(size_t)255;
        return p;
    };
    const size_t MS = (size_t)Bc * Sc;            // 8192 rows
    bf16* xb   = (bf16*)alloc(MS * Dc * 2);       // 16 MB
    bf16* WqT  = (bf16*)alloc((size_t)Dc * Dc * 2);
    bf16* WkT  = (bf16*)alloc((size_t)Dc * Dc * 2);
    bf16* WvT  = (bf16*)alloc((size_t)Dc * Dc * 2);
    bf16* WoT  = (bf16*)alloc((size_t)Dc * Dc * 2);
    bf16* WrqT = (bf16*)alloc((size_t)Dc * HRc * 2);
    bf16* WrkT = (bf16*)alloc((size_t)Dc * HRc * 2);
    float* P    = (float*)alloc(MS * Dc * 4);     // 32 MB, reused as scan output O
    float* Rbuf = (float*)alloc(MS * HRc * 4);    // 24 MB, reused as bf16 Yn
    float* Qt   = (float*)alloc(MS * Dc * 4);
    float* Kt   = (float*)alloc(MS * Dc * 4);
    float* Vt   = (float*)alloc(MS * Dc * 4);
    float* alpha = (float*)alloc((size_t)Bc * Hc * Sc * 4);
    float* beta  = (float*)alloc((size_t)Bc * Hc * Sc * 4);
    float* O = P;
    bf16* Yn = (bf16*)Rbuf;

    // prep: casts and weight transposes
    int nX = (int)(MS * Dc);
    cast_f32_bf16<<<(nX + 255) / 256, 256, 0, stream>>>(x, xb, nX);
    transpose_cast<<<Dc, 256, 0, stream>>>(Wq, WqT, Dc, Dc);
    transpose_cast<<<Dc, 256, 0, stream>>>(Wk, WkT, Dc, Dc);
    transpose_cast<<<Dc, 256, 0, stream>>>(Wv, WvT, Dc, Dc);
    transpose_cast<<<Dc, 256, 0, stream>>>(Wo, WoT, Dc, Dc);
    transpose_cast<<<HRc, 256, 0, stream>>>(Wrq, WrqT, Dc, HRc);
    transpose_cast<<<HRc, 256, 0, stream>>>(Wrk, WrkT, Dc, HRc);

    dim3 gD(Dc / 64, (unsigned)(MS / 256));    // N=1024 GEMMs
    dim3 gR(HRc / 64, (unsigned)(MS / 256));   // N=768 GEMMs
    dim3 gF(Sc, Hc, Bc);
    dim3 gS(Sc, Bc);

    // Q path
    wmma_gemm<<<gD, 256, 0, stream>>>(xb, WqT, P, (int)MS, Dc, Dc);
    wmma_gemm<<<gR, 256, 0, stream>>>(xb, WrqT, Rbuf, (int)MS, HRc, Dc);
    fuse_qkv<<<gF, 64, 0, stream>>>(P, cq, qnw, Rbuf, gs, Qt, 0);
    // K path
    wmma_gemm<<<gD, 256, 0, stream>>>(xb, WkT, P, (int)MS, Dc, Dc);
    wmma_gemm<<<gR, 256, 0, stream>>>(xb, WrkT, Rbuf, (int)MS, HRc, Dc);
    fuse_qkv<<<gF, 64, 0, stream>>>(P, ckw, knw, Rbuf, gs, Kt, 1);
    // V path
    wmma_gemm<<<gD, 256, 0, stream>>>(xb, WvT, P, (int)MS, Dc, Dc);
    fuse_qkv<<<gF, 64, 0, stream>>>(P, cv, qnw, Rbuf, gs, Vt, 2);
    // gates
    gates_kernel<<<gS, 256, 0, stream>>>(x, Wf, bfv, Wl, blv, alpha, beta);
    // sequential gated delta-rule scan (P reused as output O)
    scan_kernel<<<Bc * Hc, 512, 0, stream>>>(Kt, Vt, Qt, alpha, beta, O);
    // final rmsnorm -> bf16, then output projection
    final_norm<<<gS, 256, 0, stream>>>(O, onw, Yn);
    wmma_gemm<<<gD, 256, 0, stream>>>(Yn, WoT, out, (int)MS, Dc, Dc);
}
